// ProsodyPredictor_14190571946369
// MI455X (gfx1250) — matrix operand order repaired and checked by reading, not compiled, not run on hardware
//
#include <hip/hip_runtime.h>
#include <hip/hip_bf16.h>

typedef __attribute__((ext_vector_type(16))) _Float16 v16h;
typedef __attribute__((ext_vector_type(8)))  _Float16 v8h;
typedef __attribute__((ext_vector_type(8)))  float    v8f;

#define BB   32
#define TT   1024
#define DIN  256
#define COUT 512
#define DGRU 512
#define NGM  10
#define BT   (BB*TT)          // 32768

// ---------------- WMMA fragment helpers ----------------
// A-matrix 16x32 f16 (ISA 7.12.2): lane l holds row M = (l&15);
// K values: run of 8 at k0+(l>>4)*8 and another run of 8 at +16.
__device__ inline v16h frag_cat(v8h lo, v8h hi) {
    v16h r;
#pragma unroll
    for (int e = 0; e < 8; ++e) { r[e] = lo[e]; r[e + 8] = hi[e]; }
    return r;
}

__device__ inline v16h load_afrag(const _Float16* A, int lda, int row, int k0, int lane) {
    const _Float16* rp = A + (size_t)row * lda + k0 + ((lane >> 4) << 3);
    v8h lo = *(const v8h*)(rp);
    v8h hi = *(const v8h*)(rp + 16);
    return frag_cat(lo, hi);
}

// B-matrix 32x16 f16: lane l holds column N = (l&15); K values: run of 8
// at k0+(l>>4)*8 and another run of 8 at +16 (same as A).  B[k][n] = W[n][k]
// with W row-major [N x K] -> two 16-byte loads from row n of W.
__device__ inline v16h load_bfrag(const _Float16* W, int ldb, int nrow, int k0, int lane) {
    const _Float16* rp = W + (size_t)nrow * ldb + k0 + ((lane >> 4) << 3);
    v8h lo = *(const v8h*)(rp);
    v8h hi = *(const v8h*)(rp + 16);
    return frag_cat(lo, hi);
}

// B fragment from an LDS-staged panel sB[64][32] halves (rows = N, cols = K)
__device__ inline v16h load_bfrag_lds(const _Float16* sB, int nbase, int lane) {
    const _Float16* rp = sB + (size_t)(nbase + (lane & 15)) * 32 + ((lane >> 4) << 3);
    v8h lo = *(const v8h*)(rp);
    v8h hi = *(const v8h*)(rp + 16);
    return frag_cat(lo, hi);
}

// C/D 16x16 f32: element r of lane l -> C[m0 + r + (l>=16)*8][n0 + (l&15)]
__device__ inline void store_tile(float* C, int ldc, int m0, int n0, int lane, v8f acc) {
    int col   = n0 + (lane & 15);
    int rbase = m0 + ((lane >> 4) << 3);
#pragma unroll
    for (int r = 0; r < 8; ++r)
        C[(size_t)(rbase + r) * ldc + col] = acc[r];
}

// im2col A-fragment for conv1d (K=5, PAD=2): logical K-index = k*Cin + ci.
__device__ inline v16h load_afrag_conv(const _Float16* X, int Cin, int cshift,
                                       int m, int k0, int lane) {
    v16h r;
    int b = m >> 10;
    int t = m & (TT - 1);
    v8h zero8 = {};
#pragma unroll
    for (int h = 0; h < 2; ++h) {
        int kidx = k0 + ((lane >> 4) << 3) + h * 16;
        int kk = kidx >> cshift;
        int ci = kidx & (Cin - 1);
        int ts = t + kk - 2;
        v8h v = zero8;
        if (ts >= 0 && ts < TT)
            v = *(const v8h*)(X + ((size_t)(b * TT + ts) * Cin) + ci);
#pragma unroll
        for (int e = 0; e < 8; ++e) r[h * 8 + e] = v[e];
    }
    return r;
}

// ---------------- CDNA5 async global->LDS staging ----------------
__device__ inline void async_load_b128(unsigned lds_off, const _Float16* gptr) {
    asm volatile("global_load_async_to_lds_b128 %0, %1, off"
                 :: "v"(lds_off), "v"((unsigned long long)(size_t)gptr)
                 : "memory");
}
__device__ inline void wait_async0() {
    asm volatile("s_wait_asynccnt 0" ::: "memory");
}

__device__ inline v8f wmma_step(v16h a, v16h b, v8f c) {
    return __builtin_amdgcn_wmma_f32_16x16x32_f16(false, a, false, b, (short)0, c, false, false);
}

// ---------------- pack / convert weights ----------------
__global__ void pp_pack_kernel(const float* enc, const float* w1, const float* w2,
                               const float* wih0, const float* whh0,
                               const float* wih1, const float* whh1,
                               const float* piw, const float* sgw, const float* muw,
                               _Float16* enc_h, _Float16* w1p, _Float16* w2p,
                               _Float16* wih0p, _Float16* whh0h,
                               _Float16* wih1h, _Float16* whh1h, _Float16* Wh) {
    int tid = blockIdx.x * blockDim.x + threadIdx.x;
    int stride = gridDim.x * blockDim.x;
    for (int i = tid; i < BT * DIN; i += stride) enc_h[i] = (_Float16)enc[i];
    for (int i = tid; i < COUT * 1280; i += stride) {      // w1p[n][k*256+ci]
        int n = i / 1280, rem = i - n * 1280;
        int k = rem >> 8, ci = rem & 255;
        w1p[i] = (_Float16)w1[(size_t)n * 1280 + ci * 5 + k];
    }
    for (int i = tid; i < COUT * 2560; i += stride) {      // w2p[n][k*512+ci]
        int n = i / 2560, rem = i - n * 2560;
        int k = rem >> 9, ci = rem & 511;
        w2p[i] = (_Float16)w2[(size_t)n * 2560 + ci * 5 + k];
    }
    for (int i = tid; i < 1536 * 544; i += stride) {       // pad 516 -> 544
        int n = i / 544, c = i - n * 544;
        wih0p[i] = (c < 516) ? (_Float16)wih0[(size_t)n * 516 + c] : (_Float16)0.f;
    }
    for (int i = tid; i < 1536 * 512; i += stride) whh0h[i] = (_Float16)whh0[i];
    for (int i = tid; i < 1536 * 512; i += stride) wih1h[i] = (_Float16)wih1[i];
    for (int i = tid; i < 1536 * 512; i += stride) whh1h[i] = (_Float16)whh1[i];
    for (int i = tid; i < 96 * 1024; i += stride) {        // heads pi|sigma|mu, pad->96
        int r = i >> 10, c = i & 1023;
        float v = 0.f;
        if (r < 10)      v = piw[(size_t)r * 1024 + c];
        else if (r < 50) v = sgw[(size_t)(r - 10) * 1024 + c];
        else if (r < 90) v = muw[(size_t)(r - 50) * 1024 + c];
        Wh[i] = (_Float16)v;
    }
}

// ============ blocked WMMA GEMM: C = A * W^T, 128x64 C-block per WG =========
// 8 waves: wave = M-tile; 4 N-tiles per wave reuse one A fragment; the 64x32
// B panel is staged to LDS via global_load_async_to_lds_b128 (ASYNCcnt).
__global__ void __launch_bounds__(256)
pp_gemm_f16_blocked(const _Float16* A, int lda,
                    const _Float16* W, int ldb,
                    float* C, int ldc, int N, int K) {
    __shared__ _Float16 sB[64 * 32];
    const int tid  = threadIdx.x;
    const int lane = tid & 31;
    const int wave = tid >> 5;
    const int nblocks = N >> 6;
    const int mb = blockIdx.x / nblocks;
    const int nb = blockIdx.x - mb * nblocks;
    const int m0 = mb * 128 + wave * 16;
    const int n0 = nb * 64;

    const int srow  = tid >> 2;              // 0..63
    const int schnk = (tid & 3) << 3;        // 0,8,16,24 halves
    const unsigned lds_dst = (unsigned)(size_t)&sB[srow * 32 + schnk];
    const _Float16* gsrc = W + (size_t)(n0 + srow) * ldb + schnk;

    const int arow = m0 + (lane & 15);
    v8f acc0 = {}, acc1 = {}, acc2 = {}, acc3 = {};
    for (int k = 0; k < K; k += 32) {
        async_load_b128(lds_dst, gsrc + k);
        wait_async0();
        __syncthreads();
        v16h a  = load_afrag(A, lda, arow, k, lane);
        v16h b0 = load_bfrag_lds(sB,  0, lane);
        v16h b1 = load_bfrag_lds(sB, 16, lane);
        v16h b2 = load_bfrag_lds(sB, 32, lane);
        v16h b3 = load_bfrag_lds(sB, 48, lane);
        acc0 = wmma_step(a, b0, acc0);
        acc1 = wmma_step(a, b1, acc1);
        acc2 = wmma_step(a, b2, acc2);
        acc3 = wmma_step(a, b3, acc3);
        __syncthreads();
    }
    store_tile(C, ldc, m0, n0     , lane, acc0);
    store_tile(C, ldc, m0, n0 + 16, lane, acc1);
    store_tile(C, ldc, m0, n0 + 32, lane, acc2);
    store_tile(C, ldc, m0, n0 + 48, lane, acc3);
}

// blocked conv1d-as-GEMM: same structure, A fragments via im2col gather
__global__ void __launch_bounds__(256)
pp_gemm_conv_blocked(const _Float16* X, int Cin, int cshift,
                     const _Float16* W, int Kdim, float* C, int N) {
    __shared__ _Float16 sB[64 * 32];
    const int tid  = threadIdx.x;
    const int lane = tid & 31;
    const int wave = tid >> 5;
    const int nblocks = N >> 6;
    const int mb = blockIdx.x / nblocks;
    const int nb = blockIdx.x - mb * nblocks;
    const int m0 = mb * 128 + wave * 16;
    const int n0 = nb * 64;

    const int srow  = tid >> 2;
    const int schnk = (tid & 3) << 3;
    const unsigned lds_dst = (unsigned)(size_t)&sB[srow * 32 + schnk];
    const _Float16* gsrc = W + (size_t)(n0 + srow) * Kdim + schnk;

    const int arow = m0 + (lane & 15);
    v8f acc0 = {}, acc1 = {}, acc2 = {}, acc3 = {};
    for (int k = 0; k < Kdim; k += 32) {
        async_load_b128(lds_dst, gsrc + k);
        wait_async0();
        __syncthreads();
        v16h a  = load_afrag_conv(X, Cin, cshift, arow, k, lane);
        v16h b0 = load_bfrag_lds(sB,  0, lane);
        v16h b1 = load_bfrag_lds(sB, 16, lane);
        v16h b2 = load_bfrag_lds(sB, 32, lane);
        v16h b3 = load_bfrag_lds(sB, 48, lane);
        acc0 = wmma_step(a, b0, acc0);
        acc1 = wmma_step(a, b1, acc1);
        acc2 = wmma_step(a, b2, acc2);
        acc3 = wmma_step(a, b3, acc3);
        __syncthreads();
    }
    store_tile(C, N, m0, n0     , lane, acc0);
    store_tile(C, N, m0, n0 + 16, lane, acc1);
    store_tile(C, N, m0, n0 + 32, lane, acc2);
    store_tile(C, N, m0, n0 + 48, lane, acc3);
}

// ---------------- simple GEMM (heads, N=96 not /64) ----------------
__global__ void pp_gemm_f16_kernel(const _Float16* A, int lda,
                                   const _Float16* W, int ldb,
                                   float* C, int ldc, int M, int N, int K) {
    int lane = threadIdx.x & 31;
    int wave = threadIdx.x >> 5;
    int ntiles = N >> 4;
    int gtile = blockIdx.x * 8 + wave;
    if (gtile >= (M >> 4) * ntiles) return;
    int mt = gtile / ntiles, nt = gtile - mt * ntiles;
    int m0 = mt << 4, n0 = nt << 4;
    int arow = m0 + (lane & 15);
    int brow = n0 + (lane & 15);
    v8f acc = {};
    for (int k = 0; k < K; k += 32) {
        v16h a = load_afrag(A, lda, arow, k, lane);
        v16h b = load_bfrag(W, ldb, brow, k, lane);
        acc = wmma_step(a, b, acc);
    }
    store_tile(C, ldc, m0, n0, lane, acc);
}

// ---------------- LayerNorm(+bias) + ReLU, one wave per row (C=512) ----------
__global__ void pp_ln_relu_kernel(const float* Y, const float* bias,
                                  const float* g, const float* b,
                                  _Float16* dst1, int ld1, int off1,
                                  _Float16* dst2, int ld2, int off2, int rows) {
    int wv   = (blockIdx.x * blockDim.x + threadIdx.x) >> 5;
    int lane = threadIdx.x & 31;
    if (wv >= rows) return;
    const float* yr = Y + (size_t)wv * COUT;
    float vals[16];
    float s = 0.f;
#pragma unroll
    for (int i = 0; i < 16; ++i) { int c = lane + i * 32; vals[i] = yr[c] + bias[c]; s += vals[i]; }
    for (int m = 16; m >= 1; m >>= 1) s += __shfl_xor(s, m, 32);
    float mean = s * (1.f / COUT);
    float vs = 0.f;
#pragma unroll
    for (int i = 0; i < 16; ++i) { float d = vals[i] - mean; vs += d * d; }
    for (int m = 16; m >= 1; m >>= 1) vs += __shfl_xor(vs, m, 32);
    float inv = rsqrtf(vs * (1.f / COUT) + 1e-5f);
#pragma unroll
    for (int i = 0; i < 16; ++i) {
        int c = lane + i * 32;
        float o = (vals[i] - mean) * inv * g[c] + b[c];
        o = fmaxf(o, 0.f);
        dst1[(size_t)wv * ld1 + off1 + c] = (_Float16)o;
        if (dst2) dst2[(size_t)wv * ld2 + off2 + c] = (_Float16)o;
    }
}

// ---------------- prenet (4x4) + Xcat padding + tp passthrough --------------
__global__ void pp_prenet_kernel(const float* tp, const float* pw, const float* pb,
                                 _Float16* Xcat, float* out_tp) {
    int bt = blockIdx.x * blockDim.x + threadIdx.x;
    if (bt >= BT) return;
    int b = bt >> 10, t = bt & (TT - 1);
    float prev[4];
#pragma unroll
    for (int j = 0; j < 4; ++j)
        prev[j] = (t == 0) ? 0.f : tp[((size_t)b * TT + (t - 1)) * 4 + j];
#pragma unroll
    for (int j = 0; j < 4; ++j) {
        float s = pb[j];
#pragma unroll
        for (int i = 0; i < 4; ++i) s += prev[i] * pw[j * 4 + i];
        Xcat[(size_t)bt * 544 + 512 + j] = (_Float16)s;
    }
    for (int c = 516; c < 544; ++c) Xcat[(size_t)bt * 544 + c] = (_Float16)0.f;
#pragma unroll
    for (int j = 0; j < 4; ++j) out_tp[(size_t)bt * 4 + j] = tp[(size_t)bt * 4 + j];
}

__device__ inline float pp_sigmoid(float x) { return 1.f / (1.f + __expf(-x)); }

// ---------------- persistent 2-layer GRU+zoneout scan over T -----------------
__global__ void __launch_bounds__(1024)
pp_recurrent_kernel(const _Float16* whh0, const _Float16* wih1, const _Float16* whh1,
                    const float* bih0, const float* bhh0,
                    const float* bih1, const float* bhh1,
                    const float* gi0,                       // [BT,1536] x@wih0^T
                    float* gh0, float* gh1, float* gi1,     // [32,1536] scratch
                    float* h0f, float* h1f,                 // [32,512] f32 state
                    _Float16* h0h, _Float16* h1h,           // [32,512] f16 state
                    _Float16* hcs) {                        // [BT,1024]; h1 -> cols 0..511
    const int tid  = threadIdx.x;
    const int lane = tid & 31;
    const int wave = tid >> 5;

    for (int i = tid; i < BB * DGRU; i += 1024) {
        h0f[i] = 0.f; h1f[i] = 0.f;
        h0h[i] = (_Float16)0.f; h1h[i] = (_Float16)0.f;
    }
    __threadfence();
    __syncthreads();

    for (int t = 0; t < TT; ++t) {
        // phase 1: gh0 = h0 @ whh0^T ; gh1 = h1 @ whh1^T   (384 tiles)
        for (int ft = wave; ft < 384; ft += 32) {
            int g = ft / 192, rem = ft - g * 192;
            int mt = rem / 96, nt = rem - mt * 96;
            const _Float16* A  = g ? h1h  : h0h;
            const _Float16* Wm = g ? whh1 : whh0;
            float* Cd = g ? gh1 : gh0;
            int arow = (mt << 4) + (lane & 15);
            int brow = (nt << 4) + (lane & 15);
            v8f acc = {};
            for (int k = 0; k < DGRU; k += 32) {
                if (k + 32 < DGRU)
                    __builtin_prefetch(Wm + (size_t)brow * DGRU + k + 32, 0, 1);
                v16h a = load_afrag(A, DGRU, arow, k, lane);
                v16h b = load_bfrag(Wm, DGRU, brow, k, lane);
                acc = wmma_step(a, b, acc);
            }
            store_tile(Cd, 1536, mt << 4, nt << 4, lane, acc);
        }
        __threadfence();
        __syncthreads();

        // phase 2: GRU0 gate fusion + zoneout
        for (int i = tid; i < BB * DGRU; i += 1024) {
            int b = i >> 9, j = i & 511;
            size_t gr0 = ((size_t)b * TT + t) * 1536;
            int ghr = b * 1536;
            float r  = pp_sigmoid(gi0[gr0 + j]       + bih0[j]       + gh0[ghr + j]       + bhh0[j]);
            float z  = pp_sigmoid(gi0[gr0 + 512 + j] + bih0[512 + j] + gh0[ghr + 512 + j] + bhh0[512 + j]);
            float nn = tanhf(gi0[gr0 + 1024 + j] + bih0[1024 + j] +
                             r * (gh0[ghr + 1024 + j] + bhh0[1024 + j]));
            float hp = h0f[i];
            float hg = (1.f - z) * nn + z * hp;
            float hv = 0.1f * hp + 0.9f * hg;          // zoneout
            h0f[i] = hv; h0h[i] = (_Float16)hv;
        }
        __threadfence();
        __syncthreads();

        // phase 3: gi1 = h0_new @ wih1^T   (192 tiles)
        for (int ft = wave; ft < 192; ft += 32) {
            int mt = ft / 96, nt = ft - mt * 96;
            int arow = (mt << 4) + (lane & 15);
            int brow = (nt << 4) + (lane & 15);
            v8f acc = {};
            for (int k = 0; k < DGRU; k += 32) {
                if (k + 32 < DGRU)
                    __builtin_prefetch(wih1 + (size_t)brow * DGRU + k + 32, 0, 1);
                v16h a = load_afrag(h0h, DGRU, arow, k, lane);
                v16h b = load_bfrag(wih1, DGRU, brow, k, lane);
                acc = wmma_step(a, b, acc);
            }
            store_tile(gi1, 1536, mt << 4, nt << 4, lane, acc);
        }
        __threadfence();
        __syncthreads();

        // phase 4: GRU1 gate fusion + zoneout, emit h1 into hcs
        for (int i = tid; i < BB * DGRU; i += 1024) {
            int b = i >> 9, j = i & 511;
            int gr = b * 1536;
            float r  = pp_sigmoid(gi1[gr + j]       + bih1[j]       + gh1[gr + j]       + bhh1[j]);
            float z  = pp_sigmoid(gi1[gr + 512 + j] + bih1[512 + j] + gh1[gr + 512 + j] + bhh1[512 + j]);
            float nn = tanhf(gi1[gr + 1024 + j] + bih1[1024 + j] +
                             r * (gh1[gr + 1024 + j] + bhh1[1024 + j]));
            float hp = h1f[i];
            float hg = (1.f - z) * nn + z * hp;
            float hv = 0.1f * hp + 0.9f * hg;
            h1f[i] = hv; h1h[i] = (_Float16)hv;
            hcs[((size_t)b * TT + t) * 1024 + j] = (_Float16)hv;
        }
        __threadfence();
        __syncthreads();
    }
}

// ---------------- finalize: softmax(pi), elu(sigma)+1, mu + biases -----------
__global__ void pp_finalize_kernel(const float* H,  // [BT,96]
                                   const float* pi_b, const float* sg_b, const float* mu_b,
                                   float* out_pi, float* out_sigma, float* out_mu) {
    int bt = blockIdx.x * blockDim.x + threadIdx.x;
    if (bt >= BT) return;
    const float* hr = H + (size_t)bt * 96;
    float lg[NGM];
    float mx = -1e30f;
#pragma unroll
    for (int g = 0; g < NGM; ++g) { lg[g] = hr[g] + pi_b[g]; mx = fmaxf(mx, lg[g]); }
    float se = 0.f;
#pragma unroll
    for (int g = 0; g < NGM; ++g) { lg[g] = __expf(lg[g] - mx); se += lg[g]; }
    float inv = 1.f / se;
#pragma unroll
    for (int g = 0; g < NGM; ++g) out_pi[(size_t)bt * NGM + g] = lg[g] * inv;
#pragma unroll
    for (int i = 0; i < 40; ++i) {
        float s = hr[10 + i] + sg_b[i];
        float e = (s > 0.f) ? s : (__expf(s) - 1.f);
        out_sigma[(size_t)bt * 40 + i] = e + 1.f;
    }
#pragma unroll
    for (int i = 0; i < 40; ++i) out_mu[(size_t)bt * 40 + i] = hr[50 + i] + mu_b[i];
}

// ============================ host launcher =================================
extern "C" void kernel_launch(void* const* d_in, const int* in_sizes, int n_in,
                              void* d_out, int out_size, void* d_ws, size_t ws_size,
                              hipStream_t stream) {
    const float* enc   = (const float*)d_in[0];
    const float* tp    = (const float*)d_in[1];
    const float* w1    = (const float*)d_in[2];
    const float* b1    = (const float*)d_in[3];
    const float* ln1g  = (const float*)d_in[4];
    const float* ln1b  = (const float*)d_in[5];
    const float* w2    = (const float*)d_in[6];
    const float* b2    = (const float*)d_in[7];
    const float* ln2g  = (const float*)d_in[8];
    const float* ln2b  = (const float*)d_in[9];
    const float* pw    = (const float*)d_in[10];
    const float* pbias = (const float*)d_in[11];
    const float* wih0  = (const float*)d_in[12];
    const float* whh0  = (const float*)d_in[13];
    const float* bih0  = (const float*)d_in[14];
    const float* bhh0  = (const float*)d_in[15];
    const float* wih1  = (const float*)d_in[16];
    const float* whh1  = (const float*)d_in[17];
    const float* bih1  = (const float*)d_in[18];
    const float* bhh1  = (const float*)d_in[19];
    const float* piw   = (const float*)d_in[20];
    const float* pib   = (const float*)d_in[21];
    const float* sgw   = (const float*)d_in[22];
    const float* sgb   = (const float*)d_in[23];
    const float* muw   = (const float*)d_in[24];
    const float* mub   = (const float*)d_in[25];

    float* out = (float*)d_out;
    float* out_tp    = out;
    float* out_pi    = out + 131072;
    float* out_sigma = out + 458752;
    float* out_mu    = out + 1769472;

    char* wp = (char*)d_ws;
    auto alloc = [&](size_t bytes) -> char* {
        char* p = wp;
        wp += (bytes + 255) & ~(size_t)255;
        return p;
    };
    _Float16* enc_h  = (_Float16*)alloc((size_t)BT * DIN * 2);
    _Float16* w1p    = (_Float16*)alloc((size_t)COUT * 1280 * 2);
    _Float16* w2p    = (_Float16*)alloc((size_t)COUT * 2560 * 2);
    _Float16* wih0p  = (_Float16*)alloc((size_t)1536 * 544 * 2);
    _Float16* whh0h  = (_Float16*)alloc((size_t)1536 * 512 * 2);
    _Float16* wih1h  = (_Float16*)alloc((size_t)1536 * 512 * 2);
    _Float16* whh1h  = (_Float16*)alloc((size_t)1536 * 512 * 2);
    _Float16* Wh     = (_Float16*)alloc((size_t)96 * 1024 * 2);
    float*    convf  = (float*)   alloc((size_t)BT * COUT * 4);   // reused for heads
    _Float16* c1h    = (_Float16*)alloc((size_t)BT * COUT * 2);
    _Float16* Xcat   = (_Float16*)alloc((size_t)BT * 544 * 2);
    _Float16* hcs    = (_Float16*)alloc((size_t)BT * 1024 * 2);
    float*    gi0    = (float*)   alloc((size_t)BT * 1536 * 4);
    float*    gh0    = (float*)   alloc((size_t)BB * 1536 * 4);
    float*    gh1    = (float*)   alloc((size_t)BB * 1536 * 4);
    float*    gi1    = (float*)   alloc((size_t)BB * 1536 * 4);
    float*    h0f    = (float*)   alloc((size_t)BB * DGRU * 4);
    float*    h1f    = (float*)   alloc((size_t)BB * DGRU * 4);
    _Float16* h0h    = (_Float16*)alloc((size_t)BB * DGRU * 2);
    _Float16* h1h    = (_Float16*)alloc((size_t)BB * DGRU * 2);
    float*    headf  = convf;   // alias: conv scratch dead by head-GEMM time

    // 1) pack/convert
    pp_pack_kernel<<<512, 256, 0, stream>>>(enc, w1, w2, wih0, whh0, wih1, whh1,
                                            piw, sgw, muw,
                                            enc_h, w1p, w2p, wih0p, whh0h, wih1h, whh1h, Wh);
    // 2) conv1: M=32768, N=512, K=1280 -> 256 x 8 blocks
    pp_gemm_conv_blocked<<<2048, 256, 0, stream>>>(enc_h, DIN, 8, w1p, 1280, convf, COUT);
    // 3) LN1 + ReLU -> c1h
    pp_ln_relu_kernel<<<4096, 256, 0, stream>>>(convf, b1, ln1g, ln1b,
                                                c1h, COUT, 0, nullptr, 0, 0, BT);
    // 4) conv2: M=32768, N=512, K=2560
    pp_gemm_conv_blocked<<<2048, 256, 0, stream>>>(c1h, COUT, 9, w2p, 2560, convf, COUT);
    // 5) LN2 + ReLU -> Xcat[:,0:512] and hcs[:,512:1024]
    pp_ln_relu_kernel<<<4096, 256, 0, stream>>>(convf, b2, ln2g, ln2b,
                                                Xcat, 544, 0, hcs, 1024, 512, BT);
    // 6) prenet -> Xcat[:,512:516] (+pad), tp passthrough
    pp_prenet_kernel<<<BT / 256, 256, 0, stream>>>(tp, pw, pbias, Xcat, out_tp);
    // 7) gi0 = Xcat @ wih0^T: M=32768, N=1536, K=544 -> 256 x 24 blocks
    pp_gemm_f16_blocked<<<6144, 256, 0, stream>>>(Xcat, 544, wih0p, 544,
                                                  gi0, 1536, 1536, 544);
    // 8) sequential 2-layer GRU scan (single persistent workgroup, 32 waves)
    pp_recurrent_kernel<<<1, 1024, 0, stream>>>(whh0h, wih1h, whh1h,
                                                bih0, bhh0, bih1, bhh1,
                                                gi0, gh0, gh1, gi1,
                                                h0f, h1f, h0h, h1h, hcs);
    // 9) heads: hcs @ Wh^T (M=32768, N=96, K=1024)
    pp_gemm_f16_kernel<<<1536, 256, 0, stream>>>(hcs, 1024, Wh, 1024,
                                                 headf, 96, BT, 96, 1024);
    // 10) softmax / elu+1 / mu
    pp_finalize_kernel<<<BT / 256, 256, 0, stream>>>(headf, pib, sgb, mub,
                                                     out_pi, out_sigma, out_mu);
    (void)in_sizes; (void)n_in; (void)out_size; (void)ws_size;
}
